// NeuralALU_62380105007372
// MI455X (gfx1250) — matrix operand order repaired and checked by reading, not compiled, not run on hardware
//
#include <hip/hip_runtime.h>
#include <stdint.h>

// MI455X / gfx1250 fused NeuralALU forward, v4.
// Structure exploited:
//  - W2_N2B == I256            -> n2b output is its softmax.
//  - add-logits separable      -> exp(logit-max) = Ea[a]*Eb[b]*Ec[c]; in WMMA
//    A-layout the a-index is constant per k-step, so the exp'd A operand is
//    (scalar Ea[kk]) * (fixed per-lane v16h Ebc slice): v_pk_mul_f16 only.
//  - n2b logits separable      -> out = (Esh ⊗ Esl)/Z, pure VALU + float4 stores.
//  - Z harvested from an all-ones column of the sum|cout WMMA GEMM.
// Double-buffered LDS staging fed by global_load_async_to_lds_b128; next byte's
// loads are issued before the compute body so HBM streaming overlaps compute.
// v4: b2n A-fragments loaded as float4 (ds_load_b128 x4, single wait) before
// conversion — removes the serialized ds_load_2addr_b32 + wait chains.

typedef __attribute__((ext_vector_type(16))) _Float16 v16h;
typedef __attribute__((ext_vector_type(8)))  float    v8f;

#define NSCALE 100.0f
#define THREADS 128
#define ROWS_PER_WG 64   // 4 waves x 16 rows

__device__ __forceinline__ v8f wmma16(v16h a, v16h b, v8f c) {
  return __builtin_amdgcn_wmma_f32_16x16x32_f16(false, a, false, b, (short)0, c,
                                                false, false);
}

// A-matrix fragment (16x32) from f32 LDS rows, converted to f16.
// Vectorized: 4 x ds_load_b128, then packed f32->f16 conversions.
__device__ __forceinline__ v16h load_a_frag_f32(const float* base, int ld, int k0, int lane) {
  const int m  = lane & 15;
  const int ko = (lane < 16) ? 0 : 8;
  const float4* r = (const float4*)(base + m * ld + k0 + ko);
  const float4 x0 = r[0];
  const float4 x1 = r[1];
  const float4 y0 = r[4];   // +16 floats
  const float4 y1 = r[5];
  v16h out;
  out[0]  = (_Float16)x0.x; out[1]  = (_Float16)x0.y;
  out[2]  = (_Float16)x0.z; out[3]  = (_Float16)x0.w;
  out[4]  = (_Float16)x1.x; out[5]  = (_Float16)x1.y;
  out[6]  = (_Float16)x1.z; out[7]  = (_Float16)x1.w;
  out[8]  = (_Float16)y0.x; out[9]  = (_Float16)y0.y;
  out[10] = (_Float16)y0.z; out[11] = (_Float16)y0.w;
  out[12] = (_Float16)y1.x; out[13] = (_Float16)y1.y;
  out[14] = (_Float16)y1.z; out[15] = (_Float16)y1.w;
  return out;
}

// B-matrix fragment (32x16 f16) from a transposed [N][K] weight table.
__device__ __forceinline__ v16h load_b_fragT(const _Float16* wT, int ldk, int n0, int k0, int lane) {
  const int n  = lane & 15;
  const int ko = (lane < 16) ? 0 : 8;
  const _Float16* r = wT + (n0 + n) * ldk + k0 + ko;
  v16h out;
#pragma unroll
  for (int j = 0; j < 16; ++j) out[j] = r[(j < 8) ? j : j + 8];
  return out;
}

// C/D fragment (16x16 f32): vgpr r, lanes 0-15 -> M=r, lanes 16-31 -> M=8+r.
__device__ __forceinline__ void store_c_h(_Float16* dst, int ld, int n0, v8f c, int lane) {
  const int n  = lane & 15;
  const int mb = (lane < 16) ? 0 : 8;
#pragma unroll
  for (int r = 0; r < 8; ++r) dst[(mb + r) * ld + n0 + n] = (_Float16)c[r];
}

// ---- CDNA5 async global -> LDS (ASYNCcnt path) ----
__device__ __forceinline__ void async_b128(unsigned lds_off, const void* g) {
  asm volatile("global_load_async_to_lds_b128 %0, %1, off"
               :: "v"(lds_off), "v"((unsigned long long)(uintptr_t)g)
               : "memory");
}
__device__ __forceinline__ void wait_async0() {
  asm volatile("s_wait_asynccnt 0x0" ::: "memory");
}

__device__ __forceinline__ void issue_stage(const float* Ag, const float* Bg,
                                            float* stA, float* stB,
                                            long gbase, int tid) {
#pragma unroll 4
  for (int it = tid; it < 64 * 64; it += THREADS) {   // 64 rows x 64 b128 chunks
    int r = it >> 6, c = (it & 63) * 4;
    unsigned loff = (unsigned)((r * 256 + c) * 4);
    async_b128((unsigned)(uintptr_t)stA + loff, Ag + gbase + r * 1024 + c);
    async_b128((unsigned)(uintptr_t)stB + loff, Bg + gbase + r * 1024 + c);
  }
}

__global__ void __launch_bounds__(THREADS)
neural_alu_kernel(const float* __restrict__ Ag, const float* __restrict__ Bg,
                  float* __restrict__ Out, int Bn) {
  extern __shared__ char smem[];
  float*    stage  = (float*)smem;                  // 2 bufs x (A|B) x 64x256 f32
  float*    carryB = stage + 2 * 2 * 64 * 256;      // 4 waves x 16 x 2 f32
  _Float16* WscT   = (_Float16*)(carryB + 128);     // [32][512] (n,k) sum|cout|ones
  _Float16* Wb2nT  = WscT + 32 * 512;               // [32][256] (n,k) nibble pooling
  _Float16* bufLo  = Wb2nT + 32 * 256;              // 4 x 16 x 32: [al | bl]
  _Float16* bufHi  = bufLo + 4 * 16 * 32;           // 4 x 16 x 32: [ah | bh]
  _Float16* comb2  = bufHi + 4 * 16 * 32;           // 4 x 16 x 32: [sh | sl]

  const int tid  = threadIdx.x;
  const int lane = tid & 31;
  const int wave = tid >> 5;

  // ---- Build constant tables (once per WG) ----
  for (int t = tid; t < 32 * 512; t += THREADS) {   // WscT[n][k]
    int n = t >> 9, k = t & 511;
    int s = (k >> 5) + ((k >> 1) & 15) + (k & 1);
    float v = 0.f;
    if (n < 16)       v = ((s & 15) == n) ? 1.f : 0.f;
    else if (n == 16) v = (s <= 15) ? 1.f : 0.f;
    else if (n == 17) v = (s > 15) ? 1.f : 0.f;
    else if (n == 18) v = 1.f;                      // ones column -> row sum Z
    WscT[t] = (_Float16)v;
  }
  for (int t = tid; t < 32 * 256; t += THREADS) {   // Wb2nT[n][k]
    int n = t >> 8, k = t & 255;
    float v = (n < 16) ? (((k >> 4) == n) ? 1.f : 0.f)
                       : (((k & 15) == (n - 16)) ? 1.f : 0.f);
    Wb2nT[t] = (_Float16)v;
  }
  __syncthreads();

  const int rowBase = blockIdx.x * ROWS_PER_WG + wave * 16;
  _Float16* bLo = bufLo + wave * 16 * 32;
  _Float16* bHi = bufHi + wave * 16 * 32;
  _Float16* cb2 = comb2 + wave * 16 * 32;
  float*    cry = carryB + wave * 32;

  if (lane < 16) { cry[lane * 2 + 0] = 1.f; cry[lane * 2 + 1] = 0.f; }  // carry_zero

  const int nn = lane & 15;     // also the batch row this lane serves in A-layout
  const int ko = (lane < 16) ? 0 : 8;

  const long gbase0 = (long)blockIdx.x * ROWS_PER_WG * 1024;
  issue_stage(Ag, Bg, stage + 0 * 16384, stage + 1 * 16384, gbase0, tid);  // byte 0

  for (int byte = 0; byte < 4; ++byte) {
    wait_async0();       // own async issues done
    __syncthreads();     // everyone's done -> current buffer fully loaded
    const int  buf  = byte & 1;
    const float* sA = stage + (buf * 2 + 0) * 16384 + wave * 16 * 256;
    const float* sB = stage + (buf * 2 + 1) * 16384 + wave * 16 * 256;

    if (byte < 3) {      // prefetch next byte into the other buffer (overlaps all compute)
      const int nb = (byte + 1) & 1;
      issue_stage(Ag, Bg, stage + (nb * 2 + 0) * 16384, stage + (nb * 2 + 1) * 16384,
                  gbase0 + (byte + 1) * 256, tid);
    }

    // ---- b2n: [16,256] @ Wb2n -> high|low nibble pools (WMMA) ----
    for (int src = 0; src < 2; ++src) {
      const float* s = src ? sB : sA;
      v8f ch = {}, cl = {};
#pragma unroll
      for (int kk = 0; kk < 8; ++kk) {
        v16h af = load_a_frag_f32(s, 256, kk * 32, lane);
        ch = wmma16(af, load_b_fragT(Wb2nT, 256, 0,  kk * 32, lane), ch);
        cl = wmma16(af, load_b_fragT(Wb2nT, 256, 16, kk * 32, lane), cl);
      }
      store_c_h(bHi, 32, src * 16, ch, lane);
      store_c_h(bLo, 32, src * 16, cl, lane);
    }

    // ---- two soft nibble-adds (low then high; carry ripples) ----
    for (int half = 0; half < 2; ++half) {
      const _Float16* comb = half ? bHi : bLo;   // row m: [a-nib(16) | b-nib(16)]
      const int m = nn;

      float av[16], bv[16];
#pragma unroll
      for (int j = 0; j < 16; ++j) {
        av[j] = (float)comb[m * 32 + j];
        bv[j] = (float)comb[m * 32 + 16 + j];
      }
      const float c0v = cry[m * 2 + 0], c1v = cry[m * 2 + 1];

      float mA = av[0], mB = bv[0];
#pragma unroll
      for (int j = 1; j < 16; ++j) { mA = fmaxf(mA, av[j]); mB = fmaxf(mB, bv[j]); }
      const float mC = fmaxf(c0v, c1v);

      float Ea[16], Eb[16];
#pragma unroll
      for (int j = 0; j < 16; ++j) {
        Ea[j] = __expf((av[j] - mA) * NSCALE);
        Eb[j] = __expf((bv[j] - mB) * NSCALE);
      }
      const float Ec0 = __expf((c0v - mC) * NSCALE);
      const float Ec1 = __expf((c1v - mC) * NSCALE);

      // Fixed per-lane Ebc slice of the exp'd A operand (k within a 32-block).
      v16h ebcF;
#pragma unroll
      for (int j = 0; j < 8; ++j) {
        int k0 = ko + j, k1 = 16 + ko + j;
        ebcF[j]     = (_Float16)(Eb[(k0 >> 1) & 15] * ((k0 & 1) ? Ec1 : Ec0));
        ebcF[8 + j] = (_Float16)(Eb[(k1 >> 1) & 15] * ((k1 & 1) ? Ec1 : Ec0));
      }

      // sum|cout|Z = (Ea[kk] * ebcF) @ Wsc   (A-side is one packed f16 scale)
      v8f s0 = {}, s1 = {};
#pragma unroll 4
      for (int kk = 0; kk < 16; ++kk) {
        const _Float16 sa = (_Float16)Ea[kk];
        v16h pa;
#pragma unroll
        for (int j = 0; j < 16; ++j) pa[j] = ebcF[j] * sa;
        s0 = wmma16(pa, load_b_fragT(WscT, 512, 0,  kk * 32, lane), s0);
        s1 = wmma16(pa, load_b_fragT(WscT, 512, 16, kk * 32, lane), s1);
      }
      // Z lives in column 18 (lane 2 / lane 18); broadcast and fold 1/Z.
      const int zsrc = (lane & 16) | 2;
      v8f rz;
#pragma unroll
      for (int r = 0; r < 8; ++r) rz[r] = __builtin_amdgcn_rcpf(__shfl(s1[r], zsrc, 32));
#pragma unroll
      for (int r = 0; r < 8; ++r) { s0[r] *= rz[r]; s1[r] *= rz[r]; }

      store_c_h(cb2, 32, half ? 0 : 16, s0, lane);   // sh -> cols0-15, sl -> cols16-31
      const int mb = (lane < 16) ? 0 : 8;
      if (nn < 2)                                     // carry-out (Wsc cols 16,17)
#pragma unroll
        for (int r = 0; r < 8; ++r) cry[(mb + r) * 2 + nn] = s1[r];
    }

    // ---- n2b: out[h*16+l] = Esh[h]*Esl[l] / (sumEsh*sumEsl)  (separable) ----
    {
      const int m = nn;
      const _Float16* cr = cb2 + m * 32;             // [sh(16) | sl(16)]
      float hv[16], lv[16];
#pragma unroll
      for (int j = 0; j < 16; ++j) {
        hv[j] = (float)cr[j];
        lv[j] = (float)cr[16 + j];
      }
      float mH = hv[0], mL = lv[0];
#pragma unroll
      for (int j = 1; j < 16; ++j) { mH = fmaxf(mH, hv[j]); mL = fmaxf(mL, lv[j]); }
      float Eh[16], El[16], sH = 0.f, sL = 0.f;
#pragma unroll
      for (int j = 0; j < 16; ++j) {
        Eh[j] = __expf((hv[j] - mH) * NSCALE); sH += Eh[j];
        El[j] = __expf((lv[j] - mL) * NSCALE); sL += El[j];
      }
      const float rz = __builtin_amdgcn_rcpf(sH * sL);

      const int hbase = (lane < 16) ? 0 : 8;         // two lanes per row, 128 cols each
      float* orow = Out + ((long)(rowBase + m) * 4 + byte) * 256 + hbase * 16;
#pragma unroll
      for (int h = 0; h < 8; ++h) {
        const float eh = Eh[hbase + h] * rz;
#pragma unroll
        for (int l = 0; l < 16; l += 4) {
          float4 o;
          o.x = eh * El[l + 0];
          o.y = eh * El[l + 1];
          o.z = eh * El[l + 2];
          o.w = eh * El[l + 3];
          *(float4*)(orow + h * 16 + l) = o;
        }
      }
    }
  }
}

extern "C" void kernel_launch(void* const* d_in, const int* in_sizes, int n_in,
                              void* d_out, int out_size, void* d_ws, size_t ws_size,
                              hipStream_t stream) {
  const float* a = (const float*)d_in[0];
  const float* b = (const float*)d_in[1];
  float* out = (float*)d_out;
  const int B = in_sizes[0] / 1024;          // [B,4,256] f32
  const int grid = (B + ROWS_PER_WG - 1) / ROWS_PER_WG;
  const size_t smem = 324096;                // ~316.5 KB of the 320 KB WGP LDS
  hipFuncSetAttribute((const void*)neural_alu_kernel,
                      hipFuncAttributeMaxDynamicSharedMemorySize, (int)smem);
  neural_alu_kernel<<<grid, THREADS, smem, stream>>>(a, b, out, B);
}
